// Attention_9122510537215
// MI455X (gfx1250) — matrix operand order repaired and checked by reading, not compiled, not run on hardware
//
#include <hip/hip_runtime.h>
#include <hip/hip_bf16.h>

#define B_      4
#define LQ_     256
#define LK_     50000
#define EIN     128
#define KVIN    256
#define OUT_    256
#define NCHUNK  64
#define TK      32          // k-rows per inner iteration
#define NITER   25          // 64*25*32 = 51200 >= 50000
#define LDSW    264         // padded LDS row pitch (bf16 elems): 528B = 132 dwords -> stride-4 banks
#define NORM    0.0625f     // 1/sqrt(256)
#define NEG_INF (-1e30f)

typedef __bf16 bf16;
typedef bf16  v16bf __attribute__((ext_vector_type(16)));
typedef bf16  v8bf  __attribute__((ext_vector_type(8)));
typedef float v8f   __attribute__((ext_vector_type(8)));
typedef float v4f   __attribute__((ext_vector_type(4)));

union Frag { v16bf v; v8bf h[2]; };

// A fragment (16x32, row-major source, stride ld):
// lane<16 : row=lane, K = {0..7, 16..23}; lane>=16: row=lane-16, K = {8..15, 24..31}
__device__ __forceinline__ v16bf load_a_frag(const bf16* base, int ld) {
  const int lane = threadIdx.x & 31;
  const int r = lane & 15, hh = lane >> 4;
  const bf16* p = base + r * ld + hh * 8;
  Frag f;
  f.h[0] = *(const v8bf*)(p);
  f.h[1] = *(const v8bf*)(p + 16);
  return f.v;
}

// B fragment (32x16) where B = W^T, W row-major [N][K] stride ld:
// N = lane%16, lane<16: K = {0..7, 16..23}; lane>=16: K = {8..15, 24..31}
__device__ __forceinline__ v16bf load_bT_frag(const bf16* base, int ld) {
  const int lane = threadIdx.x & 31;
  const int n = lane & 15, hh = lane >> 4;
  const bf16* p = base + n * ld + hh * 8;
  Frag f;
  f.h[0] = *(const v8bf*)(p);
  f.h[1] = *(const v8bf*)(p + 16);
  return f.v;
}

// ---- kernel 0: weights f32 -> bf16 ----
__global__ void wconv_kernel(const float* __restrict__ Wk, const float* __restrict__ Wv,
                             bf16* __restrict__ Wkbf, bf16* __restrict__ Wvbf) {
  int i = blockIdx.x * 256 + threadIdx.x;     // 65536 elements each
  Wkbf[i] = (bf16)Wk[i];
  Wvbf[i] = (bf16)Wv[i];
}

// ---- kernel 1: Q = query @ Wq^T + bq  (tiny: 67 MFLOP, all L2-resident) ----
__global__ void qproj_kernel(const float* __restrict__ query, const float* __restrict__ Wq,
                             const float* __restrict__ bq, bf16* __restrict__ Qbf) {
  int idx = blockIdx.x * 256 + threadIdx.x;   // B*LQ*OUT = 262144
  int o  = idx & 255;
  int qq = (idx >> 8) & 255;
  int b  = idx >> 16;
  const float* x = query + ((size_t)b * LQ_ + qq) * EIN;
  const float* w = Wq + (size_t)o * EIN;
  float s = bq[o];
#pragma unroll 8
  for (int i = 0; i < EIN; i += 4) {
    v4f xv = *(const v4f*)(x + i);
    v4f wv = *(const v4f*)(w + i);
    s += xv[0]*wv[0] + xv[1]*wv[1] + xv[2]*wv[2] + xv[3]*wv[3];
  }
  Qbf[idx] = (bf16)s;   // Qbf[b][q][o], row-major in o (K-dim for the S GEMM)
}

// ---- kernel 2: fused streaming K/V projection + flash attention ----
__global__ __launch_bounds__(256)
void attn_main_kernel(const float* __restrict__ input,   // [B][LK][256] f32
                      const bf16*  __restrict__ Qbf,     // [B][256][256] bf16
                      const bf16*  __restrict__ Wkbf,    // [256][256] bf16
                      const bf16*  __restrict__ Wvbf,    // [256][256] bf16
                      const float* __restrict__ bk,
                      const float* __restrict__ bv,
                      float* __restrict__ partial)       // [B][NCHUNK][256][3]
{
  __shared__ __align__(16) bf16 Xs[TK * LDSW];
  __shared__ __align__(16) bf16 Ks[TK * LDSW];
  __shared__ __align__(16) bf16 Vs[TK * LDSW];

  const int chunk = blockIdx.x;
  const int b     = blockIdx.y;
  const int tid   = threadIdx.x;
  const int wave  = tid >> 5;
  const int lane  = tid & 31;
  const int ln    = lane & 15;
  const int hh    = lane >> 4;

  // Preload this wave's Q fragments (rows [wave*32, wave*32+32), full K=256): 16 frags, held in VGPRs.
  v16bf qf[2][8];
  {
    const bf16* Qb = Qbf + ((size_t)b * LQ_ + wave * 32) * OUT_;
#pragma unroll
    for (int mt = 0; mt < 2; ++mt)
#pragma unroll
      for (int kk = 0; kk < 8; ++kk)
        qf[mt][kk] = load_a_frag(Qb + mt * 16 * OUT_ + kk * 32, OUT_);
  }

  // Online-softmax state; row q = wave*32 + mt*16 + v + hh*8, replicated across the 16 lanes of a half.
  float mstate[2][8], lstate[2][8], astate[2][8];
#pragma unroll
  for (int mt = 0; mt < 2; ++mt)
#pragma unroll
    for (int v = 0; v < 8; ++v) { mstate[mt][v] = NEG_INF; lstate[mt][v] = 0.f; astate[mt][v] = 0.f; }

  // Bias for this wave's K/V output columns (o = (2*wave+wt)*16 + ln): constant along M -> C-operand init.
  float biask[2], biasv[2];
#pragma unroll
  for (int wt = 0; wt < 2; ++wt) {
    int o = (2 * wave + wt) * 16 + ln;
    biask[wt] = bk[o];
    biasv[wt] = bv[o];
  }

  for (int it = 0; it < NITER; ++it) {
    const long kbase = ((long)chunk * NITER + it) * TK;
    if (kbase >= LK_) break;

    // ---- Phase A: stage X [32 x 256] f32 -> bf16 LDS (zero-fill OOB rows) ----
    {
      const int r  = tid >> 3;
      const int c0 = (tid & 7) * 32;
      const long krow = kbase + r;
      const float* src = input + ((size_t)b * LK_ + krow) * KVIN + c0;
      bf16 tmp[32];
      if (krow < LK_) {
#pragma unroll
        for (int i = 0; i < 8; ++i) {
          v4f x = *(const v4f*)(src + i * 4);
#pragma unroll
          for (int j = 0; j < 4; ++j) tmp[i * 4 + j] = (bf16)x[j];
        }
        if (it + 1 < NITER && krow + TK < LK_)
          __builtin_prefetch(src + (size_t)TK * KVIN, 0, 0);   // global_prefetch_b8 next tile
      } else {
#pragma unroll
        for (int i = 0; i < 32; ++i) tmp[i] = (bf16)0.f;
      }
      bf16* dst = Xs + r * LDSW + c0;
#pragma unroll
      for (int i = 0; i < 4; ++i)
        *(v8bf*)(dst + i * 8) = *(const v8bf*)(tmp + i * 8);
    }
    __syncthreads();

    // ---- Phase B: K/V projection (wave owns o-tiles {2w, 2w+1}; shared A fragment for K and V) ----
#pragma unroll
    for (int wt = 0; wt < 2; ++wt) {
      const int nt = 2 * wave + wt;
      const bf16* Wkrow = Wkbf + (size_t)(nt * 16) * KVIN;
      const bf16* Wvrow = Wvbf + (size_t)(nt * 16) * KVIN;
#pragma unroll
      for (int mt = 0; mt < 2; ++mt) {
        v8f ak, av;
#pragma unroll
        for (int e = 0; e < 8; ++e) { ak[e] = biask[wt]; av[e] = biasv[wt]; }
#pragma unroll
        for (int kk = 0; kk < 8; ++kk) {
          v16bf a   = load_a_frag(Xs + mt * 16 * LDSW + kk * 32, LDSW);
          v16bf bkf = load_bT_frag(Wkrow + kk * 32, KVIN);
          ak = __builtin_amdgcn_wmma_f32_16x16x32_bf16(false, a, false, bkf, (short)0, ak, false, false);
          v16bf bvf = load_bT_frag(Wvrow + kk * 32, KVIN);
          av = __builtin_amdgcn_wmma_f32_16x16x32_bf16(false, a, false, bvf, (short)0, av, false, false);
        }
        const int col = nt * 16 + ln;
#pragma unroll
        for (int v = 0; v < 8; ++v) {
          const int row = mt * 16 + v + hh * 8;     // C/D layout: VGPR v -> rows v / v+8 per half
          Ks[row * LDSW + col] = (bf16)ak[v];
          Vs[row * LDSW + col] = (bf16)av[v];
        }
      }
    }
    __syncthreads();

    // ---- Phase C: S = Q @ K_tile^T for this wave's 32 q-rows (B frags = K_tile rows from LDS) ----
    v8f sacc[2][2];
#pragma unroll
    for (int mt = 0; mt < 2; ++mt)
#pragma unroll
      for (int nt = 0; nt < 2; ++nt) {
        v8f c;
#pragma unroll
        for (int e = 0; e < 8; ++e) c[e] = 0.f;
#pragma unroll
        for (int kk = 0; kk < 8; ++kk) {
          v16bf bfrag = load_bT_frag(Ks + nt * 16 * LDSW + kk * 32, LDSW);
          c = __builtin_amdgcn_wmma_f32_16x16x32_bf16(false, qf[mt][kk], false, bfrag, (short)0, c, false, false);
        }
        sacc[mt][nt] = c;
      }

    // ---- Phase D: online softmax + Sum_t p * V[t, q] ----
#pragma unroll
    for (int mt = 0; mt < 2; ++mt) {
#pragma unroll
      for (int v = 0; v < 8; ++v) {
        float z0 = sacc[mt][0][v] * NORM;            // col t = kbase + ln
        float z1 = sacc[mt][1][v] * NORM;            // col t = kbase + 16 + ln
        if (kbase + ln      >= LK_) z0 = NEG_INF;
        if (kbase + 16 + ln >= LK_) z1 = NEG_INF;
        float mx = fmaxf(z0, z1);
#pragma unroll
        for (int off = 1; off < 16; off <<= 1)
          mx = fmaxf(mx, __shfl_xor(mx, off, 32));   // stays within each 16-lane half
        const float mnew = fmaxf(mstate[mt][v], mx);
        const float p0 = __expf(z0 - mnew);
        const float p1 = __expf(z1 - mnew);
        const int q = wave * 32 + mt * 16 + v + hh * 8;
        const float vw0 = (float)Vs[ln * LDSW + q];          // V[t, q]: column read, pitch 264 -> no conflicts
        const float vw1 = (float)Vs[(16 + ln) * LDSW + q];
        float sum = p0 + p1;
        float dot = p0 * vw0 + p1 * vw1;
#pragma unroll
        for (int off = 1; off < 16; off <<= 1) {
          sum += __shfl_xor(sum, off, 32);
          dot += __shfl_xor(dot, off, 32);
        }
        const float corr = __expf(mstate[mt][v] - mnew);
        lstate[mt][v] = lstate[mt][v] * corr + sum;
        astate[mt][v] = astate[mt][v] * corr + dot;
        mstate[mt][v] = mnew;
      }
    }
    __syncthreads();   // protect Xs/Ks/Vs before next iteration overwrites them
  }

  // Emit per-chunk partials (lanes 0 and 16 carry the two row-halves' state)
  if (ln == 0) {
#pragma unroll
    for (int mt = 0; mt < 2; ++mt)
#pragma unroll
      for (int v = 0; v < 8; ++v) {
        const int q = wave * 32 + mt * 16 + v + hh * 8;
        float* p = partial + (((size_t)b * NCHUNK + chunk) * LQ_ + q) * 3;
        p[0] = mstate[mt][v];
        p[1] = lstate[mt][v];
        p[2] = astate[mt][v];
      }
  }
}

// ---- kernel 3: cross-chunk log-sum-exp merge ----
__global__ void reduce_kernel(const float* __restrict__ partial, float* __restrict__ out) {
  const int q = threadIdx.x;
  const int b = blockIdx.x;
  float M = NEG_INF, L = 0.f, A = 0.f;
  for (int c = 0; c < NCHUNK; ++c) {
    const float* p = partial + (((size_t)b * NCHUNK + c) * LQ_ + q) * 3;
    const float m = p[0], l = p[1], a = p[2];
    const float Mn = fmaxf(M, m);
    const float e0 = __expf(M - Mn), e1 = __expf(m - Mn);
    L = L * e0 + l * e1;
    A = A * e0 + a * e1;
    M = Mn;
  }
  out[(size_t)b * LQ_ + q] = A / L;
}

extern "C" void kernel_launch(void* const* d_in, const int* in_sizes, int n_in,
                              void* d_out, int out_size, void* d_ws, size_t ws_size,
                              hipStream_t stream) {
  const float* query = (const float*)d_in[0];
  const float* input = (const float*)d_in[1];
  const float* Wq    = (const float*)d_in[2];
  const float* bq    = (const float*)d_in[3];
  const float* Wk    = (const float*)d_in[4];
  const float* bk    = (const float*)d_in[5];
  const float* Wv    = (const float*)d_in[6];
  const float* bv    = (const float*)d_in[7];
  float* out = (float*)d_out;

  // workspace layout: Wkbf | Wvbf | Qbf (bf16) | partial (f32)
  bf16* Wkbf = (bf16*)d_ws;
  bf16* Wvbf = Wkbf + (size_t)OUT_ * KVIN;
  bf16* Qbf  = Wvbf + (size_t)OUT_ * KVIN;
  float* partial = (float*)(Qbf + (size_t)B_ * LQ_ * OUT_);

  wconv_kernel<<<(OUT_ * KVIN) / 256, 256, 0, stream>>>(Wk, Wv, Wkbf, Wvbf);
  qproj_kernel<<<(B_ * LQ_ * OUT_) / 256, 256, 0, stream>>>(query, Wq, bq, Qbf);
  dim3 grid(NCHUNK, B_);
  attn_main_kernel<<<grid, 256, 0, stream>>>(input, Qbf, Wkbf, Wvbf, bk, bv, partial);
  reduce_kernel<<<B_, 256, 0, stream>>>(partial, out);
}
